// Encoder_69887707840991
// MI455X (gfx1250) — compile-verified, matrix operand order checked by reading, not run on hardware
//
#include <hip/hip_runtime.h>
#include <stdint.h>
#include <math.h>

// ---------------- problem dims ----------------
#define T_ 256
#define B_ 64
#define E_ 256
#define U_ 512
#define G4 2048          // 4*U
#define KC 768           // E + U combined K
#define NKT 24           // KC / 32
#define NWG 16           // workgroups (hidden-dim slices) per direction
#define SLICE_ELEMS (24*8*32*16)   // 98304 bf16 per (dir,wg) weight slice = 768x128
#define WBYTES (SLICE_ELEMS*2)     // 196608 bytes of LDS weights

// ---------------- workspace layout (bytes) ----------------
#define OFF_EBF   ((size_t)0)
#define EBF_BYTES ((size_t)T_*B_*E_*2)            // 8 MiB
#define OFF_WFRAG (OFF_EBF + EBF_BYTES)
#define WFRAG_BYTES ((size_t)2*NWG*SLICE_ELEMS*2) // 6 MiB
#define OFF_HBUF  (OFF_WFRAG + WFRAG_BYTES)
#define HBUF_BYTES ((size_t)2*2*B_*U_*2)          // dir x parity x B x U bf16
#define OFF_BAR   (OFF_HBUF + HBUF_BYTES)

typedef __attribute__((ext_vector_type(16))) __bf16   v16bf;
typedef __attribute__((ext_vector_type(8)))  __bf16   v8bf;
typedef __attribute__((ext_vector_type(8)))  float    v8f;
typedef __attribute__((ext_vector_type(4)))  uint32_t u32x4;
typedef __attribute__((ext_vector_type(8)))  int      i32x8;

union V16u { v16bf v; v8bf h2[2]; };

// ---------------- init: zero h buffers + barrier counters ----------------
__global__ void init_zero_kernel(uint32_t* p, int n) {
    int i = blockIdx.x * 256 + threadIdx.x;
    if (i < n) p[i] = 0u;
}

// ---------------- embedding gather + f32->bf16, layout (t, b, e) ----------------
__global__ void embed_gather_kernel(const int* __restrict__ x,
                                    const float* __restrict__ emb,
                                    __bf16* __restrict__ ebf) {
    size_t i = (size_t)blockIdx.x * 256 + threadIdx.x;
    if (i >= (size_t)T_ * B_ * E_) return;
    int k = (int)(i & (E_ - 1));
    size_t tb = i >> 8;            // /E_
    int b = (int)(tb & (B_ - 1));
    int t = (int)(tb >> 6);        // /B_
    int tok = x[(size_t)b * T_ + t];
    ebf[i] = (__bf16)emb[(size_t)tok * E_ + k];
}

// ---------------- pack [W;R] (768x2048 f32) into per-(dir,wg) WMMA-B fragment order, bf16 ----------------
__global__ void pack_weights_kernel(const float* __restrict__ Wf, const float* __restrict__ Rf,
                                    const float* __restrict__ Wb, const float* __restrict__ Rb,
                                    __bf16* __restrict__ wfrag) {
    size_t i = (size_t)blockIdx.x * 256 + threadIdx.x;
    if (i >= (size_t)2 * NWG * SLICE_ELEMS) return;
    uint32_t d   = (uint32_t)(i / ((size_t)NWG * SLICE_ELEMS));
    uint32_t r   = (uint32_t)(i % ((size_t)NWG * SLICE_ELEMS));
    uint32_t wg  = r / SLICE_ELEMS;  uint32_t r2 = r % SLICE_ELEMS;
    uint32_t kt  = r2 / 4096;        uint32_t r3 = r2 % 4096;   // 4096 = 8*32*16
    uint32_t nt  = r3 / 512;         uint32_t r4 = r3 % 512;
    uint32_t lane= r4 / 16;          uint32_t e  = r4 % 16;
    uint32_t hl  = lane >> 4;        uint32_t ll = lane & 15;
    // 16-bit B-operand K pattern: lanes<16 hold K {0..7,16..23}, lanes>=16 hold K {8..15,24..31}
    uint32_t koff = e + 8u * hl + (e >= 8u ? 8u : 0u);
    uint32_t k    = kt * 32u + koff;
    uint32_t nloc = nt * 16u + ll;            // local col in [i(32)|f(32)|g(32)|o(32)]
    uint32_t gate = nloc >> 5;  uint32_t ug = nloc & 31u;
    uint32_t col  = gate * 512u + wg * 32u + ug;
    const float* W = d ? Wb : Wf;
    const float* R = d ? Rb : Rf;
    float v = (k < E_) ? W[(size_t)k * G4 + col] : R[(size_t)(k - E_) * G4 + col];
    wfrag[i] = (__bf16)v;
}

// ---------------- persistent bidirectional LSTM scan ----------------
__launch_bounds__(256, 1)
__global__ void lstm_scan_kernel(const int* __restrict__ x,
                                 const __bf16* __restrict__ ebf,
                                 const __bf16* __restrict__ wfrag,
                                 const float* __restrict__ bias_f,
                                 const float* __restrict__ bias_b,
                                 __bf16* __restrict__ hbuf,
                                 uint32_t* __restrict__ bar,
                                 float* __restrict__ out) {
    extern __shared__ char smem[];
    __bf16* w_s   = (__bf16*)smem;                 // WBYTES, fragment-linear
    float*  bias_s= (float*)(smem + WBYTES);       // 128
    float*  mask_s= bias_s + 128;                  // 64

    const int wg   = blockIdx.x;       // 0..15 hidden slice
    const int dir  = blockIdx.y;       // 0 fwd, 1 bwd
    const int tid  = threadIdx.x;
    const int lane = tid & 31, wave = tid >> 5;
    const int hl   = lane >> 4, ll = lane & 15;
    const int mtile= wave >> 1, usub = wave & 1;

    // ---- load 192 KB weight slice into LDS via Tensor Data Mover (one wave issues) ----
    if (tid < 32) {
        const __bf16* wsrc = wfrag + (size_t)(dir * NWG + wg) * SLICE_ELEMS;
        uint64_t ga = (uint64_t)(uintptr_t)wsrc;
        const uint32_t NW = WBYTES / 8;            // 24576 8-byte elements
        u32x4 g0;
        g0[0] = 1u;                                 // count = 1 (valid descriptor)
        g0[1] = 0u;                                 // lds_addr = 0 (weights first in LDS)
        g0[2] = (uint32_t)(ga & 0xFFFFFFFFull);     // global_addr[31:0]
        g0[3] = (uint32_t)((ga >> 32) & 0x1FFFFFFull) | (2u << 30); // addr[56:32] | type=2
        i32x8 g1;
        g1[0] = (int)(3u << 16);                    // data_size = 8B
        g1[1] = (int)((NW & 0xFFFFu) << 16);        // tensor_dim0[15:0]
        g1[2] = (int)(((NW >> 16) & 0xFFFFu) | (1u << 16)); // tensor_dim0 hi | tensor_dim1=1
        g1[3] = (int)((NW & 0xFFFFu) << 16);        // tile_dim0 = NW
        g1[4] = 1;                                  // tile_dim1 = 1
        g1[5] = (int)NW;                            // tensor_dim0_stride
        g1[6] = 0; g1[7] = 0;
        asm volatile("tensor_load_to_lds %0, %1" :: "s"(g0), "s"(g1) : "memory");
        __builtin_amdgcn_s_wait_tensorcnt(0);
    }
    if (tid < 128) {
        int gate = tid >> 5, ug = tid & 31;
        const float* bb = dir ? bias_b : bias_f;
        bias_s[tid] = bb[gate * 512 + wg * 32 + ug];
    }
    __syncthreads();

    const int b_a     = mtile * 16 + ll;       // A-matrix row this lane loads
    const int u_local = usub * 16 + ll;        // hidden col within 32-slice
    const int u       = wg * 32 + u_local;

    float c_reg[8], h_reg[8];
#pragma unroll
    for (int r = 0; r < 8; ++r) { c_reg[r] = 0.f; h_reg[r] = 0.f; }

    const v8f zero8 = {0.f,0.f,0.f,0.f,0.f,0.f,0.f,0.f};

    for (int s = 0; s < T_; ++s) {
        const int t   = dir ? (T_ - 1 - s) : s;
        const int par = s & 1;
        const __bf16* h_cur = hbuf + (size_t)(dir * 2 + par) * B_ * U_;
        __bf16*       h_nxt = hbuf + (size_t)(dir * 2 + (par ^ 1)) * B_ * U_;

        if (tid < B_) mask_s[tid] = (x[(size_t)tid * T_ + t] != 0) ? 1.f : 0.f;

        // prefetch next step's embedding row
        const int tn = dir ? (t > 0 ? t - 1 : 0) : (t < T_ - 1 ? t + 1 : T_ - 1);
        __builtin_prefetch(ebf + ((size_t)(tn * B_ + b_a)) * E_, 0, 1);

        v8f acc[4] = { zero8, zero8, zero8, zero8 };

        const __bf16* arow_e = ebf + ((size_t)(t * B_ + b_a)) * E_ + hl * 8;
        const __bf16* arow_h = h_cur + (size_t)b_a * U_ + hl * 8;

#pragma unroll
        for (int kt = 0; kt < NKT; ++kt) {
            const __bf16* ap = (kt < 8) ? (arow_e + kt * 32) : (arow_h + (kt - 8) * 32);
            V16u a;
            a.h2[0] = *(const v8bf*)ap;          // K base..base+7
            a.h2[1] = *(const v8bf*)(ap + 16);   // K base+16..base+23
#pragma unroll
            for (int g = 0; g < 4; ++g) {
                const int nt = g * 2 + usub;
                const __bf16* bp = w_s + ((size_t)(kt * 8 + nt) * 32 + lane) * 16;
                V16u bv;
                bv.h2[0] = *(const v8bf*)bp;
                bv.h2[1] = *(const v8bf*)(bp + 8);
                acc[g] = __builtin_amdgcn_wmma_f32_16x16x32_bf16(
                    false, a.v, false, bv.v, (short)0, acc[g], false, false);
            }
        }

        __syncthreads();   // mask_s ready for all waves

#pragma unroll
        for (int r = 0; r < 8; ++r) {
            const int b = mtile * 16 + r + 8 * hl;
            float zi = acc[0][r] + bias_s[      u_local];
            float zf = acc[1][r] + bias_s[32  + u_local];
            float zg = acc[2][r] + bias_s[64  + u_local];
            float zo = acc[3][r] + bias_s[96  + u_local];
            float iv = tanhf(zi), fv = tanhf(zf), gv = tanhf(zg), ov = tanhf(zo);
            float cn = fv * c_reg[r] + iv * gv;
            float hn = ov * tanhf(cn);
            float m  = mask_s[b];
            h_reg[r] = m * hn + (1.f - m) * h_reg[r];
            c_reg[r] = m * cn + (1.f - m) * c_reg[r];
            h_nxt[(size_t)b * U_ + u] = (__bf16)h_reg[r];
            out[((size_t)b * T_ + t) * (2 * U_) + (size_t)dir * U_ + u] = h_reg[r];
        }

        // ---- device-wide barrier across the 16 workgroups of this direction ----
        if (s != T_ - 1) {
            __threadfence();
            __syncthreads();
            if (tid == 0) {
                __hip_atomic_fetch_add(&bar[dir], 1u, __ATOMIC_RELEASE, __HIP_MEMORY_SCOPE_AGENT);
                const uint32_t target = (uint32_t)(s + 1) * NWG;
                while (__hip_atomic_load(&bar[dir], __ATOMIC_ACQUIRE, __HIP_MEMORY_SCOPE_AGENT) < target)
                    __builtin_amdgcn_s_sleep(2);
            }
            __syncthreads();
            __threadfence();
        }
    }

    // backward final hidden state -> second output
    if (dir == 1) {
        float* out2 = out + (size_t)B_ * T_ * (2 * U_);
#pragma unroll
        for (int r = 0; r < 8; ++r) {
            const int b = mtile * 16 + r + 8 * hl;
            out2[(size_t)b * U_ + u] = h_reg[r];
        }
    }
}

// ---------------- host launcher ----------------
extern "C" void kernel_launch(void* const* d_in, const int* in_sizes, int n_in,
                              void* d_out, int out_size, void* d_ws, size_t ws_size,
                              hipStream_t stream) {
    const int*   x   = (const int*)d_in[0];
    const float* emb = (const float*)d_in[1];
    const float* Wf  = (const float*)d_in[2];
    const float* Rf  = (const float*)d_in[3];
    const float* bf  = (const float*)d_in[4];
    const float* Wb  = (const float*)d_in[5];
    const float* Rb  = (const float*)d_in[6];
    const float* bb  = (const float*)d_in[7];
    float* out = (float*)d_out;
    char*  ws  = (char*)d_ws;

    __bf16*   ebf   = (__bf16*)(ws + OFF_EBF);
    __bf16*   wfrag = (__bf16*)(ws + OFF_WFRAG);
    __bf16*   hbuf  = (__bf16*)(ws + OFF_HBUF);
    uint32_t* bar   = (uint32_t*)(ws + OFF_BAR);

    // zero h double-buffers + barrier counters (deterministic per launch)
    {
        int nwords = (int)((HBUF_BYTES + 256) / 4);
        init_zero_kernel<<<(nwords + 255) / 256, 256, 0, stream>>>((uint32_t*)(ws + OFF_HBUF), nwords);
    }
    {
        size_t n = (size_t)T_ * B_ * E_;
        embed_gather_kernel<<<(unsigned)((n + 255) / 256), 256, 0, stream>>>(x, emb, ebf);
    }
    {
        size_t n = (size_t)2 * NWG * SLICE_ELEMS;
        pack_weights_kernel<<<(unsigned)((n + 255) / 256), 256, 0, stream>>>(Wf, Rf, Wb, Rb, wfrag);
    }

    const size_t smem_bytes = WBYTES + 128 * sizeof(float) + 64 * sizeof(float);
    (void)hipFuncSetAttribute((const void*)lstm_scan_kernel,
                              hipFuncAttributeMaxDynamicSharedMemorySize, (int)smem_bytes);
    lstm_scan_kernel<<<dim3(NWG, 2), 256, smem_bytes, stream>>>(
        x, ebf, wfrag, bf, bb, hbuf, bar, out);
}
